// CoherenceNetWithGCN_46858093199644
// MI455X (gfx1250) — matrix operand order repaired and checked by reading, not compiled, not run on hardware
//
#include <hip/hip_runtime.h>

// ---------------------------------------------------------------------------
// CDNA5 (gfx1250, wave32) implementation of CoherenceNetWithGCN.
// All matmuls run through V_WMMA_F32_16X16X32_BF16 (bf16 A/B, fp32 accum).
// ---------------------------------------------------------------------------

typedef __attribute__((ext_vector_type(16))) __bf16 v16bf;
typedef __attribute__((ext_vector_type(8)))  float  v8f;
typedef unsigned short u16;

__device__ __forceinline__ u16 f2bfbits(float f) {
  union { float f; unsigned u; } c; c.f = f;
  unsigned r = c.u + 0x7FFFu + ((c.u >> 16) & 1u);   // round-to-nearest-even
  return (u16)(r >> 16);
}
__device__ __forceinline__ float bf2f(u16 h) {
  union { unsigned u; float f; } c; c.u = ((unsigned)h) << 16; return c.f;
}

// ---------------------------------------------------------------------------
// Embedding gather + fp32->bf16 convert with K padding (zeros in pad region)
// ---------------------------------------------------------------------------
__global__ void gather_rows_kernel(const float* __restrict__ table,
                                   const int* __restrict__ ids,
                                   u16* __restrict__ out, int N, int D, int Kp) {
  int idx = blockIdx.x * blockDim.x + threadIdx.x;
  if (idx >= N * Kp) return;
  int r = idx / Kp, c = idx - r * Kp;
  float v = (c < D) ? table[(long long)ids[r] * D + c] : 0.0f;
  out[idx] = f2bfbits(v);
}

// fp32 [K,N] -> bf16 [Kp,N], zero rows K..Kp-1
__global__ void cvt_pad_kernel(const float* __restrict__ src, u16* __restrict__ dst,
                               int K, int Kp, int N) {
  int idx = blockIdx.x * blockDim.x + threadIdx.x;
  if (idx >= Kp * N) return;
  int r = idx / N;
  dst[idx] = f2bfbits(r < K ? src[idx] : 0.0f);
}

// bf16 row gather (attender = stmts[cand_ids])
__global__ void gather_bf16_rows_kernel(const u16* __restrict__ src,
                                        const int* __restrict__ ids,
                                        u16* __restrict__ dst, int Nr, int D) {
  int idx = blockIdx.x * blockDim.x + threadIdx.x;
  if (idx >= Nr * D) return;
  int r = idx / D, c = idx - r * D;
  dst[idx] = src[(long long)ids[r] * D + c];
}

// ---------------------------------------------------------------------------
// WMMA GEMM: C[M,N] (+)= A[M,K] * B[K,N]
//   A: bf16 (A_F32=false) or fp32 converted on the fly (A_F32=true)
//   A_TRANS: element (m,k) at A[k*lda + m]  (for adj^T and w^T GEMMs)
//   B: bf16 row-major [K,N]; staged TRANSPOSED into LDS so each lane's
//      fragment is contiguous (collapses to ds_load_b128 pairs).
// Block: 128 threads = 4 waves; tile 64x128; each wave owns a 32x64 strip
// = 2 A-frags x 4 B-frags = 8 WMMA per K-step. K-step = 32.
// ---------------------------------------------------------------------------
#define TM 64
#define TN 128
#define TKS 32

union Frag { v16bf v; unsigned u32[8]; u16 h[16]; };

template<bool A_F32, bool A_TRANS>
__global__ __launch_bounds__(128) void gemm_wmma(const void* __restrict__ Ap, int lda,
    const u16* __restrict__ B, int ldb,
    float* __restrict__ C, int ldc, int M, int N, int K, int acc) {
  __shared__ u16 As[TM][TKS + 8];
  __shared__ u16 Bs[TN][TKS + 8];   // transposed: Bs[n][k]

  const int tid  = threadIdx.x;
  const int lane = tid & 31;
  const int wave = tid >> 5;
  const int m0 = blockIdx.y * TM;
  const int n0 = blockIdx.x * TN;
  const int mq  = (wave >> 1) * 32;  // wave row offset in tile (0/32)
  const int nqw = (wave & 1) * 64;   // wave col offset in tile (0/64)

  const float* Af = (const float*)Ap;
  const u16*   Ah = (const u16*)Ap;

  // C fragment mapping (16x16 f32 layout): VGPR r -> M=r (lanes0-15) / M=8+r
  const int l15 = lane & 15;
  const int mhl = (lane >> 4) << 3;

  v8f accv[2][4] = {};
  if (acc) {
#pragma unroll
    for (int i = 0; i < 2; ++i) {
      int mbase = m0 + mq + i * 16 + mhl;
#pragma unroll
      for (int j = 0; j < 4; ++j) {
        int nn = n0 + nqw + j * 16 + l15;
#pragma unroll
        for (int r = 0; r < 8; ++r)
          accv[i][j][r] = C[(long long)(mbase + r) * ldc + nn];
      }
    }
  }

  for (int k0 = 0; k0 < K; k0 += TKS) {
    // prefetch next K-tile (lowers to global_prefetch_b8 on gfx1250)
    if (k0 + TKS < K) {
      if (A_TRANS) __builtin_prefetch(&Af[(long long)(k0 + TKS) * lda + m0], 0, 1);
      else if (A_F32) __builtin_prefetch(&Af[(long long)(m0 + (tid >> 1)) * lda + k0 + TKS], 0, 1);
      else __builtin_prefetch(&Ah[(long long)(m0 + (tid >> 1)) * lda + k0 + TKS], 0, 1);
      __builtin_prefetch(&B[(long long)(k0 + TKS + (tid >> 2)) * ldb + n0], 0, 1);
    }

    // ---- stage A tile: As[m][k] ----
    if (A_TRANS) {
      for (int idx = tid; idx < TM * TKS; idx += 128) {
        int kk = idx / TM, ma = idx - kk * TM;   // consecutive tid -> consecutive m (coalesced)
        float v = A_F32 ? Af[(long long)(k0 + kk) * lda + (m0 + ma)]
                        : bf2f(Ah[(long long)(k0 + kk) * lda + (m0 + ma)]);
        As[ma][kk] = f2bfbits(v);
      }
    } else {
      for (int idx = tid; idx < TM * (TKS / 2); idx += 128) {
        int rr = idx / (TKS / 2), cc = (idx - rr * (TKS / 2)) * 2;
        unsigned pack;
        if (A_F32) {
          const float* p = &Af[(long long)(m0 + rr) * lda + (k0 + cc)];
          pack = (unsigned)f2bfbits(p[0]) | ((unsigned)f2bfbits(p[1]) << 16);
        } else {
          pack = *(const unsigned*)&Ah[(long long)(m0 + rr) * lda + (k0 + cc)];
        }
        *(unsigned*)&As[rr][cc] = pack;
      }
    }
    // ---- stage B tile transposed: Bs[n][k] ----
    // Read two adjacent K-rows (coalesced over n), repack into [n][k] pairs.
    for (int idx = tid; idx < (TKS / 2) * (TN / 2); idx += 128) {
      int kk = idx / (TN / 2);
      int nn2 = (idx - kk * (TN / 2)) * 2;
      unsigned r0 = *(const unsigned*)&B[(long long)(k0 + 2 * kk)     * ldb + n0 + nn2];
      unsigned r1 = *(const unsigned*)&B[(long long)(k0 + 2 * kk + 1) * ldb + n0 + nn2];
      *(unsigned*)&Bs[nn2][2 * kk]     = (r0 & 0xffffu) | (r1 << 16);
      *(unsigned*)&Bs[nn2 + 1][2 * kk] = (r0 >> 16) | (r1 & 0xffff0000u);
    }
    __syncthreads();

    // ---- build fragments (all contiguous -> ds_load_b128 pairs) ----
    // A 16x32: lanes0-15 row M=lane K={0..7,16..23}; lanes16-31 K={8..15,24..31}
    Frag a[2], b[4];
    {
      const int kb = (lane < 16) ? 0 : 8;
#pragma unroll
      for (int i = 0; i < 2; ++i) {
        const u16* row = &As[mq + i * 16 + l15][0];
#pragma unroll
        for (int v = 0; v < 4; ++v)
          a[i].u32[v] = *(const unsigned*)&row[kb + 2 * v];
#pragma unroll
        for (int v = 4; v < 8; ++v)
          a[i].u32[v] = *(const unsigned*)&row[16 + kb + 2 * (v - 4)];
      }
      // B 32x16: lanes0-15 col N=lane K={0..7,16..23}; lanes16-31 col N=lane-16 K={8..15,24..31}
      const int kb2 = (lane < 16) ? 0 : 8;
#pragma unroll
      for (int j = 0; j < 4; ++j) {
        const u16* col = &Bs[nqw + j * 16 + l15][kb2];
#pragma unroll
        for (int v = 0; v < 8; ++v)
          b[j].u32[v] = *(const unsigned*)&col[v < 4 ? 2 * v : 16 + 2 * (v - 4)];
      }
    }

#pragma unroll
    for (int i = 0; i < 2; ++i)
#pragma unroll
      for (int j = 0; j < 4; ++j)
        accv[i][j] = __builtin_amdgcn_wmma_f32_16x16x32_bf16(
            false, a[i].v, false, b[j].v, (short)0, accv[i][j], false, false);

    __syncthreads();
  }

#pragma unroll
  for (int i = 0; i < 2; ++i) {
    int mbase = m0 + mq + i * 16 + mhl;
#pragma unroll
    for (int j = 0; j < 4; ++j) {
      int nn = n0 + nqw + j * 16 + l15;
#pragma unroll
      for (int r = 0; r < 8; ++r)
        C[(long long)(mbase + r) * ldc + nn] = accv[i][j][r];
    }
  }
}

// ---------------------------------------------------------------------------
// Epilogue: bias + activation (0=none, 1=relu, 2=tanh), write bf16
// ---------------------------------------------------------------------------
__global__ void epilogue_kernel(const float* __restrict__ Cin, const float* __restrict__ bias,
                                int M, int N, int act, u16* __restrict__ outb) {
  int idx = blockIdx.x * blockDim.x + threadIdx.x;
  if (idx >= M * N) return;
  int n = idx % N;
  float v = Cin[idx] + (bias ? bias[n] : 0.0f);
  if (act == 1) v = fmaxf(v, 0.0f);
  else if (act == 2) v = tanhf(v);
  outb[idx] = f2bfbits(v);
}

// ---------------------------------------------------------------------------
// Attention score: out[i,j] = sum_h tanh(a[i,h] + bv[j,h] + bc[h]) * Ws[h] + bs
// One wave per row i; each lane owns 16 of the 512 H elements; shfl reduce.
// ---------------------------------------------------------------------------
__global__ __launch_bounds__(256) void score_kernel(
    const float* __restrict__ a, const float* __restrict__ bv,
    const float* __restrict__ bc, const float* __restrict__ Ws,
    const float* __restrict__ bs, float* __restrict__ out, int Na) {
  const int lane = threadIdx.x & 31;
  const int wave = threadIdx.x >> 5;
  const int i = blockIdx.x * (blockDim.x >> 5) + wave;
  if (i >= Na) return;
  const int h0 = lane * 16;
  float av[16], wv[16], bcv[16];
#pragma unroll
  for (int t = 0; t < 16; ++t) {
    av[t]  = a[(long long)i * 512 + h0 + t];
    wv[t]  = Ws[h0 + t];
    bcv[t] = bc[h0 + t];
  }
  const float bsv = bs[0];
  for (int j = 0; j < 64; ++j) {
    float s = 0.0f;
#pragma unroll
    for (int t = 0; t < 16; ++t)
      s += tanhf(av[t] + bv[j * 512 + h0 + t] + bcv[t]) * wv[t];
#pragma unroll
    for (int off = 16; off > 0; off >>= 1) s += __shfl_xor(s, off, 32);
    if (lane == 0) out[i * 64 + j] = s + bsv;
  }
}

// Column-wise softmax over axis 0 of w[Na,64]; one block per column j.
__global__ __launch_bounds__(256) void softmax_col_kernel(float* __restrict__ w, int Na) {
  const int j = blockIdx.x, t = threadIdx.x;
  __shared__ float red[256];
  float m = -3.0e38f;
  for (int i = t; i < Na; i += 256) m = fmaxf(m, w[i * 64 + j]);
  red[t] = m; __syncthreads();
  for (int s = 128; s > 0; s >>= 1) { if (t < s) red[t] = fmaxf(red[t], red[t + s]); __syncthreads(); }
  const float mx = red[0]; __syncthreads();
  float sum = 0.0f;
  for (int i = t; i < Na; i += 256) sum += __expf(w[i * 64 + j] - mx);
  red[t] = sum; __syncthreads();
  for (int s = 128; s > 0; s >>= 1) { if (t < s) red[t] += red[t + s]; __syncthreads(); }
  const float inv = 1.0f / red[0];
  __syncthreads();
  for (int i = t; i < Na; i += 256) w[i * 64 + j] = __expf(w[i * 64 + j] - mx) * inv;
}

// Final: out[j] = dot(att_bf[j,:], W_coh) + b_coh ; one wave per j
__global__ void final_dot_kernel(const u16* __restrict__ att, const float* __restrict__ Wcoh,
                                 const float* __restrict__ bcoh, float* __restrict__ out) {
  const int j = blockIdx.x, lane = threadIdx.x;
  float s = 0.0f;
  for (int t = lane; t < 512; t += 32) s += bf2f(att[j * 512 + t]) * Wcoh[t];
#pragma unroll
  for (int off = 16; off > 0; off >>= 1) s += __shfl_xor(s, off, 32);
  if (lane == 0) out[j] = s + bcoh[0];
}

// ---------------------------------------------------------------------------
// Host orchestration
// ---------------------------------------------------------------------------
extern "C" void kernel_launch(void* const* d_in, const int* in_sizes, int n_in,
                              void* d_out, int out_size, void* d_ws, size_t ws_size,
                              hipStream_t stream) {
  (void)in_sizes; (void)n_in; (void)out_size; (void)ws_size;
  constexpr int NE = 2048, NS = 4096, NC = 64, DE = 300, H = 512, KP = 320;

  const float* adj        = (const float*)d_in[0];
  const int*   ere_ids    = (const int*)d_in[1];
  const int*   stmt_ids   = (const int*)d_in[2];
  const int*   cand_ids   = (const int*)d_in[3];
  const float* ere_table  = (const float*)d_in[4];
  const float* stmt_table = (const float*)d_in[5];
  const float* W_ere_init = (const float*)d_in[6];
  const float* b_ere_init = (const float*)d_in[7];
  const float* W_stmt_init= (const float*)d_in[8];
  const float* b_stmt_init= (const float*)d_in[9];
  const float* W_s2e_init = (const float*)d_in[10];
  const float* b_s2e_init = (const float*)d_in[11];
  const float* W_e2s_init = (const float*)d_in[12];
  const float* b_e2s_init = (const float*)d_in[13];
  const float* W_ere      = (const float*)d_in[14];
  const float* b_ere      = (const float*)d_in[15];
  const float* W_stmt     = (const float*)d_in[16];
  const float* b_stmt     = (const float*)d_in[17];
  const float* W_s2e      = (const float*)d_in[18];
  const float* b_s2e      = (const float*)d_in[19];
  const float* W_e2s      = (const float*)d_in[20];
  const float* b_e2s      = (const float*)d_in[21];
  const float* Wc_ss      = (const float*)d_in[22];
  const float* bc_ss      = (const float*)d_in[23];
  const float* Ws_ss      = (const float*)d_in[24];
  const float* bs_ss      = (const float*)d_in[25];
  const float* Wc_es      = (const float*)d_in[26];
  const float* bc_es      = (const float*)d_in[27];
  const float* Ws_es      = (const float*)d_in[28];
  const float* bs_es      = (const float*)d_in[29];
  const float* W_att      = (const float*)d_in[30];
  const float* b_att      = (const float*)d_in[31];
  const float* W_coh      = (const float*)d_in[32];
  const float* b_coh      = (const float*)d_in[33];
  float* out = (float*)d_out;

  char* ws = (char*)d_ws;
  size_t off = 0;
  auto alloc = [&](size_t bytes) -> char* {
    off = (off + 255) & ~(size_t)255;
    char* p = ws + off; off += bytes; return p;
  };

  // bf16 operand buffers
  u16* Ebf          = (u16*)alloc((size_t)NE * KP * 2);
  u16* Sbf          = (u16*)alloc((size_t)NS * KP * 2);
  u16* Wb_ere_init  = (u16*)alloc((size_t)KP * H * 2);
  u16* Wb_stmt_init = (u16*)alloc((size_t)KP * H * 2);
  u16* Wb_s2e_init  = (u16*)alloc((size_t)3 * KP * H * 2);
  u16* Wb_e2s_init  = (u16*)alloc((size_t)2 * H * H * 2);
  u16* Wb_ere       = (u16*)alloc((size_t)H * H * 2);
  u16* Wb_stmt      = (u16*)alloc((size_t)H * H * 2);
  u16* Wb_s2e       = (u16*)alloc((size_t)3 * H * H * 2);
  u16* Wb_e2s       = (u16*)alloc((size_t)2 * H * H * 2);
  u16* Wb_cs        = (u16*)alloc((size_t)2 * H * H * 2);   // Wc_ss [1024,512]
  u16* Wb_ces       = (u16*)alloc((size_t)2 * H * H * 2);   // Wc_es
  u16* Wb_att       = (u16*)alloc((size_t)3 * H * H * 2);   // W_att [1536,512]
  u16* s2e_bf       = (u16*)alloc((size_t)3 * NS * H * 2);
  u16* e2s_bf       = (u16*)alloc((size_t)2 * NE * H * 2);
  u16* eres_bf1     = (u16*)alloc((size_t)NE * H * 2);
  u16* eres_bf2     = (u16*)alloc((size_t)NE * H * 2);
  u16* stmts_bf1    = (u16*)alloc((size_t)NS * H * 2);
  u16* stmts_bf2    = (u16*)alloc((size_t)NS * H * 2);
  u16* attender_bf  = (u16*)alloc((size_t)NC * H * 2);
  u16* att_bf       = (u16*)alloc((size_t)NC * H * 2);
  // fp32 scratch
  float* Cstmt   = (float*)alloc((size_t)NS * H * 4);
  float* Ceres   = (float*)alloc((size_t)NE * H * 4);
  float* Cbb_ss  = (float*)alloc((size_t)NC * H * 4);
  float* Cbb_es  = (float*)alloc((size_t)NC * H * 4);
  float* Wss     = (float*)alloc((size_t)NS * NC * 4);
  float* Wes     = (float*)alloc((size_t)NE * NC * 4);
  float* Cctx_ss = (float*)alloc((size_t)NC * H * 4);
  float* Cctx_es = (float*)alloc((size_t)NC * H * 4);
  float* Catt    = (float*)alloc((size_t)NC * H * 4);

  auto cvt = [&](const float* src, u16* dst, int K, int Kp, int N_) {
    int total = Kp * N_;
    cvt_pad_kernel<<<(total + 255) / 256, 256, 0, stream>>>(src, dst, K, Kp, N_);
  };
  auto gemmNN = [&](const u16* A, int lda, const u16* B, int ldb, float* C, int ldc,
                    int M_, int N_, int K_, int acc) {
    gemm_wmma<false, false><<<dim3(N_ / TN, M_ / TM), 128, 0, stream>>>(
        (const void*)A, lda, B, ldb, C, ldc, M_, N_, K_, acc);
  };
  auto gemmFN = [&](const float* A, int lda, const u16* B, int ldb, float* C, int ldc,
                    int M_, int N_, int K_, int acc) {
    gemm_wmma<true, false><<<dim3(N_ / TN, M_ / TM), 128, 0, stream>>>(
        (const void*)A, lda, B, ldb, C, ldc, M_, N_, K_, acc);
  };
  auto gemmFT = [&](const float* A, int lda, const u16* B, int ldb, float* C, int ldc,
                    int M_, int N_, int K_, int acc) {
    gemm_wmma<true, true><<<dim3(N_ / TN, M_ / TM), 128, 0, stream>>>(
        (const void*)A, lda, B, ldb, C, ldc, M_, N_, K_, acc);
  };
  auto epi = [&](const float* Cin, const float* bias, int M_, int N_, int act, u16* ob) {
    int total = M_ * N_;
    epilogue_kernel<<<(total + 255) / 256, 256, 0, stream>>>(Cin, bias, M_, N_, act, ob);
  };

  // ---- precision conversions ----
  cvt(W_ere_init, Wb_ere_init, DE, KP, H);
  cvt(W_stmt_init, Wb_stmt_init, DE, KP, H);
  for (int r = 0; r < 3; ++r) cvt(W_s2e_init + (size_t)r * DE * H, Wb_s2e_init + (size_t)r * KP * H, DE, KP, H);
  cvt(W_e2s_init, Wb_e2s_init, 2 * H, 2 * H, H);
  cvt(W_ere, Wb_ere, H, H, H);
  cvt(W_stmt, Wb_stmt, H, H, H);
  cvt(W_s2e, Wb_s2e, 3 * H, 3 * H, H);
  cvt(W_e2s, Wb_e2s, 2 * H, 2 * H, H);
  cvt(Wc_ss, Wb_cs, 2 * H, 2 * H, H);
  cvt(Wc_es, Wb_ces, 2 * H, 2 * H, H);
  cvt(W_att, Wb_att, 3 * H, 3 * H, H);
  gather_rows_kernel<<<(NE * KP + 255) / 256, 256, 0, stream>>>(ere_table, ere_ids, Ebf, NE, DE, KP);
  gather_rows_kernel<<<(NS * KP + 255) / 256, 256, 0, stream>>>(stmt_table, stmt_ids, Sbf, NS, DE, KP);

  // ---- GCN layer 1 ----
  for (int r = 0; r < 3; ++r) {
    gemmNN(Sbf, KP, Wb_s2e_init + (size_t)r * KP * H, H, Cstmt, H, NS, H, KP, 0);
    epi(Cstmt, b_s2e_init + r * H, NS, H, 0, s2e_bf + (size_t)r * NS * H);
  }
  gemmNN(Ebf, KP, Wb_ere_init, H, Ceres, H, NE, H, KP, 0);
  for (int r = 0; r < 3; ++r)
    gemmFN(adj + (size_t)r * NE * NS, NS, s2e_bf + (size_t)r * NS * H, H, Ceres, H, NE, H, NS, 1);
  epi(Ceres, b_ere_init, NE, H, 1, eres_bf1);

  for (int r = 0; r < 2; ++r) {
    gemmNN(eres_bf1, H, Wb_e2s_init + (size_t)r * H * H, H, Ceres, H, NE, H, H, 0);
    epi(Ceres, b_e2s_init + r * H, NE, H, 0, e2s_bf + (size_t)r * NE * H);
  }
  gemmNN(Sbf, KP, Wb_stmt_init, H, Cstmt, H, NS, H, KP, 0);
  for (int r = 0; r < 2; ++r)
    gemmFT(adj + (size_t)r * NE * NS, NS, e2s_bf + (size_t)r * NE * H, H, Cstmt, H, NS, H, NE, 1);
  epi(Cstmt, b_stmt_init, NS, H, 1, stmts_bf1);

  // ---- GCN layer 2 ----
  for (int r = 0; r < 3; ++r) {
    gemmNN(stmts_bf1, H, Wb_s2e + (size_t)r * H * H, H, Cstmt, H, NS, H, H, 0);
    epi(Cstmt, b_s2e + r * H, NS, H, 0, s2e_bf + (size_t)r * NS * H);
  }
  gemmNN(eres_bf1, H, Wb_ere, H, Ceres, H, NE, H, H, 0);
  for (int r = 0; r < 3; ++r)
    gemmFN(adj + (size_t)r * NE * NS, NS, s2e_bf + (size_t)r * NS * H, H, Ceres, H, NE, H, NS, 1);
  epi(Ceres, b_ere, NE, H, 1, eres_bf2);

  for (int r = 0; r < 2; ++r) {
    gemmNN(eres_bf2, H, Wb_e2s + (size_t)r * H * H, H, Ceres, H, NE, H, H, 0);
    epi(Ceres, b_e2s + r * H, NE, H, 0, e2s_bf + (size_t)r * NE * H);
  }
  gemmNN(stmts_bf1, H, Wb_stmt, H, Cstmt, H, NS, H, H, 0);
  for (int r = 0; r < 2; ++r)
    gemmFT(adj + (size_t)r * NE * NS, NS, e2s_bf + (size_t)r * NE * H, H, Cstmt, H, NS, H, NE, 1);
  epi(Cstmt, b_stmt, NS, H, 1, stmts_bf2);

  // ---- coherence attention ----
  gather_bf16_rows_kernel<<<(NC * H + 255) / 256, 256, 0, stream>>>(stmts_bf2, cand_ids, attender_bf, NC, H);

  // a = attendees @ Wc[:H] ; bv = attender @ Wc[H:]
  gemmNN(stmts_bf2, H, Wb_cs, H, Cstmt, H, NS, H, H, 0);                 // a_ss -> Cstmt
  gemmNN(eres_bf2, H, Wb_ces, H, Ceres, H, NE, H, H, 0);                 // a_es -> Ceres
  gemmNN(attender_bf, H, Wb_cs + (size_t)H * H, H, Cbb_ss, H, NC, H, H, 0);
  gemmNN(attender_bf, H, Wb_ces + (size_t)H * H, H, Cbb_es, H, NC, H, H, 0);

  score_kernel<<<NS / 8, 256, 0, stream>>>(Cstmt, Cbb_ss, bc_ss, Ws_ss, bs_ss, Wss, NS);
  score_kernel<<<NE / 8, 256, 0, stream>>>(Ceres, Cbb_es, bc_es, Ws_es, bs_es, Wes, NE);
  softmax_col_kernel<<<NC, 256, 0, stream>>>(Wss, NS);
  softmax_col_kernel<<<NC, 256, 0, stream>>>(Wes, NE);

  // ctx = w^T @ X  (A = w^T via transposed fp32 staging)
  gemmFT(Wss, NC, stmts_bf2, H, Cctx_ss, H, NC, H, NS, 0);
  gemmFT(Wes, NC, eres_bf2, H, Cctx_es, H, NC, H, NE, 0);

  // att = tanh([attender, ctx_ss, ctx_es] @ W_att + b_att)
  gemmNN(attender_bf, H, Wb_att, H, Catt, H, NC, H, H, 0);
  gemmFN(Cctx_ss, H, Wb_att + (size_t)H * H, H, Catt, H, NC, H, H, 1);
  gemmFN(Cctx_es, H, Wb_att + (size_t)2 * H * H, H, Catt, H, NC, H, H, 1);
  epi(Catt, b_att, NC, H, 2, att_bf);

  final_dot_kernel<<<NC, 32, 0, stream>>>(att_bf, W_coh, b_coh, out);
}
